// Encoder_77318001262793
// MI455X (gfx1250) — matrix operand order repaired and checked by reading, not compiled, not run on hardware
//
#include <hip/hip_runtime.h>
#include <hip/hip_bf16.h>
#include <math.h>

// ---------------- problem constants ----------------
static const int kB  = 2;
static const int kN  = 2048;
static const int kNR = 256;
static const int kDA = 256;
static const int kE  = 768;
static const int kH  = 12;
static const int kDH = 64;
static const int kROWS = kB * kN;          // 4096
static const float kLN1E4 = 9.210340371976184f; // ln(10000)

typedef __attribute__((ext_vector_type(16))) _Float16 v16h;
typedef __attribute__((ext_vector_type(8)))  float    v8f;

struct F16Frag { union { v16h v; uint4 q[2]; }; };

// =====================================================================
// Generic f16 WMMA GEMM:  C[M,Nc] = A[M,Kp] * B[Kp,Nc]  (+bias, epilogue)
// Block: 128 threads = 4 waves in 2x2; each wave computes 32x32 (2x2 WMMA).
// A row-major f16 (ld = Kp, Kp % 32 == 0, M % 64 == 0).
// Bt = B TRANSPOSED, row-major [Nc][Kp]  (pre-transposed at weight conversion)
//      so both tiles stage to LDS with straight b128 copies.
// modes: 0 = f32 store, 1 = f16 store, 2 = gelu->f16, 3 = out32 += gate * (acc+bias)
// =====================================================================
__global__ void __launch_bounds__(128)
gemm_f16(const _Float16* __restrict__ A, const _Float16* __restrict__ Bt,
         int M, int Nc, int Kp,
         const float* __restrict__ bias,
         float* __restrict__ out32, _Float16* __restrict__ out16, int ldOut,
         const float* __restrict__ gate, int gateStride, int rowsPerBatch,
         int mode)
{
    __shared__ _Float16 As[64 * 48];   // [m][k], stride 48 halves (96B; 16B-aligned chunks)
    __shared__ _Float16 Bs[64 * 48];   // [n][k]

    const int tid  = threadIdx.x;
    const int wave = tid >> 5;
    const int lane = tid & 31;
    const int waveM = wave >> 1, waveN = wave & 1;
    const int mBase = blockIdx.x * 64;
    const int nBase = blockIdx.y * 64;
    const int lrow = lane & 15, lh = lane >> 4;

    v8f acc[2][2] = {};

    for (int k0 = 0; k0 < Kp; k0 += 32) {
        // --- stage A tile 64x32 and Bt tile 64x32 (256 b128 chunks each) ---
        #pragma unroll
        for (int c = tid; c < 256; c += 128) {
            int r = c >> 2, kc = (c & 3) << 3;
            uint4 da = *(const uint4*)(A  + (size_t)(mBase + r) * Kp + k0 + kc);
            uint4 db = *(const uint4*)(Bt + (size_t)(nBase + r) * Kp + k0 + kc);
            *(uint4*)(&As[r * 48 + kc]) = da;
            *(uint4*)(&Bs[r * 48 + kc]) = db;
        }
        // prefetch next K tile into cache hierarchy (global_prefetch_b8)
        if (k0 + 32 < Kp) {
            __builtin_prefetch(A  + (size_t)(mBase + (tid >> 1)) * Kp + k0 + 32 + (tid & 1) * 16, 0, 1);
            __builtin_prefetch(Bt + (size_t)(nBase + (tid >> 1)) * Kp + k0 + 32 + (tid & 1) * 16, 0, 1);
        }
        __syncthreads();

        F16Frag afr[2], bfr[2];
        #pragma unroll
        for (int mt = 0; mt < 2; mt++) {
            const _Float16* p = &As[(waveM * 32 + mt * 16 + lrow) * 48 + lh * 8];
            afr[mt].q[0] = *(const uint4*)(p);
            afr[mt].q[1] = *(const uint4*)(p + 16);
        }
        #pragma unroll
        for (int nt = 0; nt < 2; nt++) {
            const _Float16* p = &Bs[(waveN * 32 + nt * 16 + lrow) * 48 + lh * 8];
            bfr[nt].q[0] = *(const uint4*)(p);
            bfr[nt].q[1] = *(const uint4*)(p + 16);
        }
        #pragma unroll
        for (int mt = 0; mt < 2; mt++)
            #pragma unroll
            for (int nt = 0; nt < 2; nt++)
                acc[mt][nt] = __builtin_amdgcn_wmma_f32_16x16x32_f16(
                    false, afr[mt].v, false, bfr[nt].v, (short)0, acc[mt][nt], false, false);
        __syncthreads();
    }

    // --- epilogue (M % 64 == 0: no row guard) ---
    #pragma unroll
    for (int mt = 0; mt < 2; mt++)
        #pragma unroll
        for (int nt = 0; nt < 2; nt++) {
            int gn = nBase + waveN * 32 + nt * 16 + lrow;
            #pragma unroll
            for (int r = 0; r < 8; r++) {
                int gm = mBase + waveM * 32 + mt * 16 + lh * 8 + r;
                float val = acc[mt][nt][r];
                if (bias) val += bias[gn];
                if (mode == 0) {
                    out32[(size_t)gm * ldOut + gn] = val;
                } else if (mode == 1) {
                    out16[(size_t)gm * ldOut + gn] = (_Float16)val;
                } else if (mode == 2) {
                    float t3 = val * val * val;
                    float gl = 0.5f * val * (1.f + tanhf(0.7978845608f * (val + 0.044715f * t3)));
                    out16[(size_t)gm * ldOut + gn] = (_Float16)gl;
                } else {
                    float gt = gate[(gm / rowsPerBatch) * gateStride + gn];
                    out32[(size_t)gm * ldOut + gn] += gt * val;
                }
            }
        }
}

// =====================================================================
// Fused local window attention. One wave per (b, h, qblock of 32).
// Q pre-scaled by 1/sqrt(DH). Window: [max(0,32q-48), min(N,32q+80)).
// Of layout: [B*N, 768] row-major, col = h*64 + d  (head-concat for o-proj).
// =====================================================================
__global__ void __launch_bounds__(32)
attn_local(const _Float16* __restrict__ Qh, const _Float16* __restrict__ Kh,
           const _Float16* __restrict__ Vh, _Float16* __restrict__ Of)
{
    const int lane = threadIdx.x;
    const int idx  = blockIdx.x;
    const int b  = idx / (kH * (kN / 32));
    const int rm = idx % (kH * (kN / 32));
    const int h  = rm / (kN / 32);
    const int qb = rm % (kN / 32);
    int lo = 32 * qb - 48; if (lo < 0) lo = 0;
    int hi = 32 * qb + 80; if (hi > kN) hi = kN;
    const int w = hi - lo;

    __shared__ _Float16 Qt[32 * 72];      // Q tile [32][64], stride 72
    __shared__ _Float16 KVt[128 * 72];    // K tile [128][64]; later V^T [64][136]
    __shared__ float    Sf[32 * 128];
    __shared__ _Float16 Pt[32 * 136];

    const size_t headBase = ((size_t)(b * kH + h)) * kN * kDH;
    const int lrow = lane & 15, lh = lane >> 4;

    // load Q tile
    for (int c = lane; c < 256; c += 32) {
        int r = c >> 3, kc = (c & 7) << 3;
        uint4 d = *(const uint4*)(Qh + headBase + (size_t)(qb * 32 + r) * kDH + kc);
        *(uint4*)(&Qt[r * 72 + kc]) = d;
    }
    // load K tile (rows = key slots)
    for (int c = lane; c < 1024; c += 32) {
        int r = c >> 3, kc = (c & 7) << 3;
        uint4 d; d.x = d.y = d.z = d.w = 0u;
        if (r < w) d = *(const uint4*)(Kh + headBase + (size_t)(lo + r) * kDH + kc);
        *(uint4*)(&KVt[r * 72 + kc]) = d;
    }
    __syncthreads();

    // S = Q * K^T : M=32, N=128, K=64
    v8f accS[2][8] = {};
    #pragma unroll
    for (int ks = 0; ks < 64; ks += 32) {
        F16Frag afr[2];
        #pragma unroll
        for (int mt = 0; mt < 2; mt++) {
            const _Float16* p = &Qt[(mt * 16 + lrow) * 72 + ks + lh * 8];
            afr[mt].q[0] = *(const uint4*)(p);
            afr[mt].q[1] = *(const uint4*)(p + 16);
        }
        #pragma unroll
        for (int nt = 0; nt < 8; nt++) {
            F16Frag bfr;
            const _Float16* p = &KVt[(nt * 16 + lrow) * 72 + ks + lh * 8];
            bfr.q[0] = *(const uint4*)(p);
            bfr.q[1] = *(const uint4*)(p + 16);
            #pragma unroll
            for (int mt = 0; mt < 2; mt++)
                accS[mt][nt] = __builtin_amdgcn_wmma_f32_16x16x32_f16(
                    false, afr[mt].v, false, bfr.v, (short)0, accS[mt][nt], false, false);
        }
    }
    #pragma unroll
    for (int mt = 0; mt < 2; mt++)
        #pragma unroll
        for (int nt = 0; nt < 8; nt++)
            #pragma unroll
            for (int r = 0; r < 8; r++)
                Sf[(mt * 16 + lh * 8 + r) * 128 + nt * 16 + lrow] = accS[mt][nt][r];
    __syncthreads();

    // softmax: one row per lane
    {
        float mx = -1e30f;
        for (int j = 0; j < w; j++) mx = fmaxf(mx, Sf[lane * 128 + j]);
        float sum = 0.f;
        for (int j = 0; j < w; j++) {
            float e = __expf(Sf[lane * 128 + j] - mx);
            Sf[lane * 128 + j] = e; sum += e;
        }
        float inv = 1.f / sum;
        for (int j = 0; j < w; j++)   Pt[lane * 136 + j] = (_Float16)(Sf[lane * 128 + j] * inv);
        for (int j = w; j < 128; j++) Pt[lane * 136 + j] = (_Float16)0.f;
    }
    __syncthreads();

    // load V transposed into KVt as Vt[d][key], stride 136 (8704 <= 9216 halves)
    for (int c = lane; c < 1024; c += 32) {
        int r = c >> 3, kc = (c & 7) << 3;
        union { uint4 q; _Float16 e[8]; } u; u.q.x = u.q.y = u.q.z = u.q.w = 0u;
        if (r < w) u.q = *(const uint4*)(Vh + headBase + (size_t)(lo + r) * kDH + kc);
        #pragma unroll
        for (int i = 0; i < 8; i++) KVt[(kc + i) * 136 + r] = u.e[i];
    }
    __syncthreads();

    // O = P * V : M=32, N=64, K=128
    v8f accO[2][4] = {};
    #pragma unroll
    for (int ks = 0; ks < 128; ks += 32) {
        F16Frag afr[2];
        #pragma unroll
        for (int mt = 0; mt < 2; mt++) {
            const _Float16* p = &Pt[(mt * 16 + lrow) * 136 + ks + lh * 8];
            afr[mt].q[0] = *(const uint4*)(p);
            afr[mt].q[1] = *(const uint4*)(p + 16);
        }
        #pragma unroll
        for (int nt = 0; nt < 4; nt++) {
            F16Frag bfr;
            const _Float16* p = &KVt[(nt * 16 + lrow) * 136 + ks + lh * 8];
            bfr.q[0] = *(const uint4*)(p);
            bfr.q[1] = *(const uint4*)(p + 16);
            #pragma unroll
            for (int mt = 0; mt < 2; mt++)
                accO[mt][nt] = __builtin_amdgcn_wmma_f32_16x16x32_f16(
                    false, afr[mt].v, false, bfr.v, (short)0, accO[mt][nt], false, false);
        }
    }
    #pragma unroll
    for (int mt = 0; mt < 2; mt++)
        #pragma unroll
        for (int nt = 0; nt < 4; nt++)
            #pragma unroll
            for (int r = 0; r < 8; r++) {
                int row = b * kN + qb * 32 + mt * 16 + lh * 8 + r;
                int col = h * kDH + nt * 16 + lrow;
                Of[(size_t)row * kE + col] = (_Float16)accO[mt][nt][r];
            }
}

// =====================================================================
// LayerNorm over rows (optional affine, optional SiLU), f32 and/or f16 out
// =====================================================================
__global__ void __launch_bounds__(256)
ln_rows(const float* __restrict__ in, int D,
        const float* __restrict__ g, const float* __restrict__ bta,
        float* __restrict__ out32, _Float16* __restrict__ out16, int ldOut, int act)
{
    const int row = blockIdx.x;
    const float* x = in + (size_t)row * D;
    __shared__ float red[256];
    const int t = threadIdx.x;
    float s = 0.f;
    for (int i = t; i < D; i += 256) s += x[i];
    red[t] = s; __syncthreads();
    for (int o = 128; o > 0; o >>= 1) { if (t < o) red[t] += red[t + o]; __syncthreads(); }
    float mu = red[0] / D; __syncthreads();
    float v = 0.f;
    for (int i = t; i < D; i += 256) { float d = x[i] - mu; v += d * d; }
    red[t] = v; __syncthreads();
    for (int o = 128; o > 0; o >>= 1) { if (t < o) red[t] += red[t + o]; __syncthreads(); }
    float rstd = rsqrtf(red[0] / D + 1e-5f);
    for (int i = t; i < D; i += 256) {
        float y = (x[i] - mu) * rstd;
        if (g) y = y * g[i] + bta[i];
        if (act) y = y / (1.f + __expf(-y));
        if (out32) out32[(size_t)row * ldOut + i] = y;
        if (out16) out16[(size_t)row * ldOut + i] = (_Float16)y;
    }
}

// adaLN modulation: out = LN(x) * (1 + sc) + sh  -> f16 (D = 768)
__global__ void __launch_bounds__(256)
mod_ln(const float* __restrict__ x, const float* __restrict__ mvec,
       int shOff, int scOff, _Float16* __restrict__ out)
{
    const int row = blockIdx.x;
    const int b = row / kN;
    const float* xr = x + (size_t)row * kE;
    __shared__ float red[256];
    const int t = threadIdx.x;
    float s = 0.f;
    for (int i = t; i < kE; i += 256) s += xr[i];
    red[t] = s; __syncthreads();
    for (int o = 128; o > 0; o >>= 1) { if (t < o) red[t] += red[t + o]; __syncthreads(); }
    float mu = red[0] / kE; __syncthreads();
    float v = 0.f;
    for (int i = t; i < kE; i += 256) { float d = xr[i] - mu; v += d * d; }
    red[t] = v; __syncthreads();
    for (int o = 128; o > 0; o >>= 1) { if (t < o) red[t] += red[t + o]; __syncthreads(); }
    float rstd = rsqrtf(red[0] / kE + 1e-5f);
    for (int i = t; i < kE; i += 256) {
        float sc = mvec[b * 4608 + scOff + i];
        float sh = mvec[b * 4608 + shOff + i];
        float y = (xr[i] - mu) * rstd * (1.f + sc) + sh;
        out[(size_t)row * kE + i] = (_Float16)y;
    }
}

// =====================================================================
// qkv bias already applied by GEMM. Apply RoPE to q,k; scale q; split heads.
// QKV f32 [4096, 2304] with col = plane*768 + h*64 + d. Out f16 [B,H,N,DH].
// =====================================================================
__global__ void qkv_rope(const float* __restrict__ QKV, const int* __restrict__ rope_pos,
                         _Float16* __restrict__ Qh, _Float16* __restrict__ Kh,
                         _Float16* __restrict__ Vh)
{
    const int row = blockIdx.x;               // 0..4095
    const int b = row / kN, n = row % kN;
    const float pos = (float)rope_pos[row];
    const float* base = QKV + (size_t)row * 2304;
    for (int p = threadIdx.x; p < kE; p += blockDim.x) {
        int h = p >> 6, d = p & 63;
        size_t oidx = ((size_t)(b * kH + h) * kN + n) * kDH + d;
        Vh[oidx] = (_Float16)base[1536 + p];
        int d0 = d & 31;
        float inv = __expf(-(float)d0 * (kLN1E4 / 32.f));
        float ang = pos * inv;
        float cs = __cosf(ang), sn = __sinf(ang);
        float q1 = base[h * 64 + d0],        q2 = base[h * 64 + d0 + 32];
        float k1 = base[768 + h * 64 + d0],  k2 = base[768 + h * 64 + d0 + 32];
        float qo = (d < 32) ? (q1 * cs - q2 * sn) : (q1 * sn + q2 * cs);
        float ko = (d < 32) ? (k1 * cs - k2 * sn) : (k1 * sn + k2 * cs);
        Qh[oidx] = (_Float16)(qo * 0.125f);   // 1/sqrt(64)
        Kh[oidx] = (_Float16)ko;
    }
}

// =====================================================================
// Build padded f16 feature rows: F[4096,928] and coord-fourier C[4096,416]
// =====================================================================
__global__ void build_feats(const float* __restrict__ ref_pos, const float* __restrict__ coords,
                            const int* __restrict__ res_id, const float* __restrict__ res_type,
                            const float* __restrict__ charge, const float* __restrict__ atomic_num,
                            const float* __restrict__ atom_name,
                            _Float16* __restrict__ F, _Float16* __restrict__ C)
{
    const int a = blockIdx.x;
    float rp0 = ref_pos[a * 3], rp1 = ref_pos[a * 3 + 1], rp2 = ref_pos[a * 3 + 2];
    float cc0 = coords[a * 3],  cc1 = coords[a * 3 + 1],  cc2 = coords[a * 3 + 2];
    float posr = (float)res_id[a];
    for (int c = threadIdx.x; c < 928; c += blockDim.x) {
        float v = 0.f;
        if (c < 3) v = (c == 0) ? rp0 : (c == 1) ? rp1 : rp2;
        else if (c < 195) {
            int s = c - 3; int f = s / 3, ax = s % 3;
            float xv = (ax == 0) ? rp0 : (ax == 1) ? rp1 : rp2;
            v = sinf(xv * exp2f(12.f * (float)f / 63.f));
        } else if (c < 387) {
            int s = c - 195; int f = s / 3, ax = s % 3;
            float xv = (ax == 0) ? rp0 : (ax == 1) ? rp1 : rp2;
            v = cosf(xv * exp2f(12.f * (float)f / 63.f));
        } else if (c < 410) v = res_type[a * 23 + (c - 387)];
        else if (c == 410) v = posr;
        else if (c < 475) { int j = c - 411; v = sinf(posr * __expf(-(2.f * j) * kLN1E4 / 128.f)); }
        else if (c < 539) { int j = c - 475; v = cosf(posr * __expf(-(2.f * j) * kLN1E4 / 128.f)); }
        else if (c == 539) v = charge[a];
        else if (c < 668) v = atomic_num[a * 128 + (c - 540)];
        else if (c < 924) v = atom_name[a * 256 + (c - 668)];
        F[(size_t)a * 928 + c] = (_Float16)v;
    }
    for (int c = threadIdx.x; c < 416; c += blockDim.x) {
        float v = 0.f;
        if (c < 3) v = (c == 0) ? cc0 : (c == 1) ? cc1 : cc2;
        else if (c < 195) {
            int s = c - 3; int f = s / 3, ax = s % 3;
            float xv = (ax == 0) ? cc0 : (ax == 1) ? cc1 : cc2;
            v = sinf(xv * exp2f(12.f * (float)f / 63.f));
        } else if (c < 387) {
            int s = c - 195; int f = s / 3, ax = s % 3;
            float xv = (ax == 0) ? cc0 : (ax == 1) ? cc1 : cc2;
            v = cosf(xv * exp2f(12.f * (float)f / 63.f));
        }
        C[(size_t)a * 416 + c] = (_Float16)v;
    }
}

// ---------------- small glue kernels ----------------
// f32 [K,Nc] -> f16 TRANSPOSED [Nc,Kp], zero-padded K
__global__ void cvt_pad_t(const float* __restrict__ src, _Float16* __restrict__ dst,
                          int K, int Kp, int Nc)
{
    size_t idx = (size_t)blockIdx.x * blockDim.x + threadIdx.x;
    if (idx >= (size_t)Kp * Nc) return;
    int n = (int)(idx / Kp), k = (int)(idx % Kp);
    dst[idx] = (k < K) ? (_Float16)src[(size_t)k * Nc + n] : (_Float16)0.f;
}

__global__ void cvt_f16(const float* __restrict__ src, _Float16* __restrict__ dst, size_t n)
{
    size_t i = (size_t)blockIdx.x * blockDim.x + threadIdx.x;
    if (i < n) dst[i] = (_Float16)src[i];
}

__global__ void small_linear(const float* __restrict__ in, const float* __restrict__ W,
                             const float* __restrict__ bias, float* __restrict__ out,
                             int Bn, int K, int Nout)
{
    int idx = blockIdx.x * blockDim.x + threadIdx.x;
    if (idx >= Bn * Nout) return;
    int b = idx / Nout, j = idx - b * Nout;
    float s = bias ? bias[j] : 0.f;
    for (int k = 0; k < K; k++) s += in[b * K + k] * W[(size_t)k * Nout + j];
    out[idx] = s;
}

__global__ void silu_vec(const float* __restrict__ in, float* __restrict__ out, int n)
{
    int i = blockIdx.x * blockDim.x + threadIdx.x;
    if (i < n) { float v = in[i]; out[i] = v / (1.f + __expf(-v)); }
}

__global__ void colsum_kernel(const float* __restrict__ a2t, float* __restrict__ cs)
{
    int idx = blockIdx.x * blockDim.x + threadIdx.x;
    if (idx >= kB * kNR) return;
    int b = idx / kNR, r = idx % kNR;
    float s = 0.f;
    for (int n = 0; n < kN; n++) s += a2t[((size_t)b * kN + n) * kNR + r];
    cs[idx] = s;
}

__global__ void residue_reduce(const float* __restrict__ a2t, const float* __restrict__ AL,
                               const float* __restrict__ cs, float* __restrict__ out)
{
    int br = blockIdx.x;                 // 0..511
    int b = br / kNR, r = br % kNR;
    int d = threadIdx.x;                 // 256
    float acc = 0.f;
    for (int n = 0; n < kN; n++) {
        float wv = a2t[((size_t)b * kN + n) * kNR + r];
        acc += wv * AL[((size_t)b * kN + n) * kDA + d];
    }
    out[(size_t)br * kDA + d] = acc / (cs[br] + 1e-6f);
}

// =====================================================================
extern "C" void kernel_launch(void* const* d_in, const int* in_sizes, int n_in,
                              void* d_out, int out_size, void* d_ws, size_t ws_size,
                              hipStream_t stream)
{
    (void)in_sizes; (void)n_in; (void)out_size; (void)ws_size;
    const float* atom_coords = (const float*)d_in[0];
    const float* ref_pos     = (const float*)d_in[1];
    const int*   res_id      = (const int*)d_in[2];
    const float* res_type    = (const float*)d_in[3];
    const float* charge      = (const float*)d_in[4];
    const float* atomic_num  = (const float*)d_in[5];
    const float* atom_name   = (const float*)d_in[6];
    const float* adaLN       = (const float*)d_in[7];
    const int*   rope_pos    = (const int*)d_in[8];
    const float* a2t         = (const float*)d_in[9];
    const float* feat_W = (const float*)d_in[10]; const float* feat_b = (const float*)d_in[11];
    const float* feat_ln_g = (const float*)d_in[12]; const float* feat_ln_b = (const float*)d_in[13];
    const float* pos_W = (const float*)d_in[14];
    const float* in_W  = (const float*)d_in[15];
    const float* tin_W = (const float*)d_in[16]; const float* tin_b = (const float*)d_in[17];
    const float* tin_ln_g = (const float*)d_in[18]; const float* tin_ln_b = (const float*)d_in[19];
    const float* ada_W = (const float*)d_in[20]; const float* ada_b = (const float*)d_in[21];
    const float* ada_ln_g = (const float*)d_in[22]; const float* ada_ln_b = (const float*)d_in[23];
    const float* mod_W = (const float*)d_in[24]; const float* mod_b = (const float*)d_in[25];
    const float* qkv_W = (const float*)d_in[26]; const float* qkv_b = (const float*)d_in[27];
    const float* o_W   = (const float*)d_in[28]; const float* o_b   = (const float*)d_in[29];
    const float* m1_W  = (const float*)d_in[30]; const float* m1_b  = (const float*)d_in[31];
    const float* m2_W  = (const float*)d_in[32]; const float* m2_b  = (const float*)d_in[33];
    const float* tout_W = (const float*)d_in[34]; const float* tout_b = (const float*)d_in[35];
    const float* tout_ln_g = (const float*)d_in[36]; const float* tout_ln_b = (const float*)d_in[37];

    char* ws = (char*)d_ws;
    size_t off = 0;
    auto a16 = [&](size_t n) -> _Float16* {
        _Float16* p = (_Float16*)(ws + off); off += ((n * 2 + 255) / 256) * 256; return p; };
    auto a32 = [&](size_t n) -> float* {
        float* p = (float*)(ws + off); off += ((n * 4 + 255) / 256) * 256; return p; };

    // f16 weights, TRANSPOSED [Nout][Kp], K padded to multiple of 32
    _Float16* wFeat = a16((size_t)928 * 256);
    _Float16* wPos  = a16((size_t)416 * 256);
    _Float16* wIn   = a16((size_t)512 * 256);
    _Float16* wTin  = a16((size_t)256 * 768);
    _Float16* wQkv  = a16((size_t)2 * 768 * 2304);
    _Float16* wO    = a16((size_t)2 * 768 * 768);
    _Float16* wM1   = a16((size_t)2 * 768 * 3072);
    _Float16* wM2   = a16((size_t)2 * 3072 * 768);
    _Float16* wTout = a16((size_t)768 * 256);
    // activations
    _Float16* F16b = a16((size_t)kROWS * 928);
    _Float16* C16b = a16((size_t)kROWS * 416);
    _Float16* XIN  = a16((size_t)kROWS * 512);
    _Float16* AIN  = a16((size_t)kROWS * 256);
    _Float16* Hmod = a16((size_t)kROWS * 768);
    _Float16* Ofb  = a16((size_t)kROWS * 768);
    _Float16* H3   = a16((size_t)kROWS * 3072);
    size_t headsz = (size_t)kB * kH * kN * kDH;
    _Float16* Qh = a16(headsz); _Float16* Kh = a16(headsz); _Float16* Vh = a16(headsz);
    float* S32 = a32((size_t)kROWS * 2304);
    float* xbuf = a32((size_t)kROWS * 768);
    float* cpre = a32(2 * 768); float* cvecf = a32(2 * 768); float* siluc = a32(2 * 768);
    float* mvec = a32(2 * 4608); float* csum = a32(kB * kNR);

    auto cld = [](size_t a, size_t b) { return (unsigned)((a + b - 1) / b); };

    // ---- weight conversion (f32 [K,N] -> f16 transposed [N,Kp]) ----
    cvt_pad_t<<<cld((size_t)928 * 256, 256), 256, 0, stream>>>(feat_W, wFeat, 924, 928, 256);
    cvt_pad_t<<<cld((size_t)416 * 256, 256), 256, 0, stream>>>(pos_W, wPos, 387, 416, 256);
    cvt_pad_t<<<cld((size_t)512 * 256, 256), 256, 0, stream>>>(in_W, wIn, 512, 512, 256);
    cvt_pad_t<<<cld((size_t)256 * 768, 256), 256, 0, stream>>>(tin_W, wTin, 256, 256, 768);
    for (int l = 0; l < 2; l++) {
        cvt_pad_t<<<cld((size_t)768 * 2304, 256), 256, 0, stream>>>(qkv_W + (size_t)l * 768 * 2304, wQkv + (size_t)l * 768 * 2304, 768, 768, 2304);
        cvt_pad_t<<<cld((size_t)768 * 768, 256), 256, 0, stream>>>(o_W + (size_t)l * 768 * 768, wO + (size_t)l * 768 * 768, 768, 768, 768);
        cvt_pad_t<<<cld((size_t)768 * 3072, 256), 256, 0, stream>>>(m1_W + (size_t)l * 768 * 3072, wM1 + (size_t)l * 768 * 3072, 768, 768, 3072);
        cvt_pad_t<<<cld((size_t)3072 * 768, 256), 256, 0, stream>>>(m2_W + (size_t)l * 3072 * 768, wM2 + (size_t)l * 3072 * 768, 3072, 3072, 768);
    }
    cvt_pad_t<<<cld((size_t)768 * 256, 256), 256, 0, stream>>>(tout_W, wTout, 768, 768, 256);

    // ---- featurize ----
    build_feats<<<kROWS, 128, 0, stream>>>(ref_pos, atom_coords, res_id, res_type, charge,
                                           atomic_num, atom_name, F16b, C16b);

    // feats path: F @ feat_W + b -> LN+SiLU -> XIN[:, :256]
    gemm_f16<<<dim3(kROWS / 64, 256 / 64), 128, 0, stream>>>(F16b, wFeat, kROWS, 256, 928,
        feat_b, S32, nullptr, 256, nullptr, 0, 0, 0);
    ln_rows<<<kROWS, 256, 0, stream>>>(S32, 256, feat_ln_g, feat_ln_b, nullptr, XIN, 512, 1);
    // coords path: C @ pos_W -> XIN[:, 256:512]
    gemm_f16<<<dim3(kROWS / 64, 256 / 64), 128, 0, stream>>>(C16b, wPos, kROWS, 256, 416,
        nullptr, nullptr, XIN + 256, 512, nullptr, 0, 0, 1);
    // atom_in = XIN @ in_W -> AIN (f16)
    gemm_f16<<<dim3(kROWS / 64, 256 / 64), 128, 0, stream>>>(XIN, wIn, kROWS, 256, 512,
        nullptr, nullptr, AIN, 256, nullptr, 0, 0, 1);
    // x = LN(AIN @ tin_W + tin_b)
    gemm_f16<<<dim3(kROWS / 64, 768 / 64), 128, 0, stream>>>(AIN, wTin, kROWS, 768, 256,
        tin_b, S32, nullptr, 768, nullptr, 0, 0, 0);
    ln_rows<<<kROWS, 256, 0, stream>>>(S32, 768, tin_ln_g, tin_ln_b, xbuf, nullptr, 768, 0);

    // ---- adaLN conditioning ----
    small_linear<<<cld(2 * 768, 256), 256, 0, stream>>>(adaLN, ada_W, ada_b, cpre, 2, 256, 768);
    ln_rows<<<2, 256, 0, stream>>>(cpre, 768, ada_ln_g, ada_ln_b, cvecf, nullptr, 768, 0);
    silu_vec<<<cld(2 * 768, 256), 256, 0, stream>>>(cvecf, siluc, 2 * 768);

    // ---- transformer blocks ----
    for (int l = 0; l < 2; l++) {
        small_linear<<<cld(2 * 4608, 256), 256, 0, stream>>>(siluc, mod_W + (size_t)l * 768 * 4608,
            mod_b + (size_t)l * 4608, mvec, 2, 768, 4608);
        // h = LN(x)*(1+sc1)+sh1
        mod_ln<<<kROWS, 256, 0, stream>>>(xbuf, mvec, 0, 768, Hmod);
        // qkv
        gemm_f16<<<dim3(kROWS / 64, 2304 / 64), 128, 0, stream>>>(Hmod, wQkv + (size_t)l * 768 * 2304,
            kROWS, 2304, 768, qkv_b + (size_t)l * 2304, S32, nullptr, 2304, nullptr, 0, 0, 0);
        qkv_rope<<<kROWS, 256, 0, stream>>>(S32, rope_pos, Qh, Kh, Vh);
        attn_local<<<kB * kH * (kN / 32), 32, 0, stream>>>(Qh, Kh, Vh, Ofb);
        // x += g1 * (O @ o_W + b)
        gemm_f16<<<dim3(kROWS / 64, 768 / 64), 128, 0, stream>>>(Ofb, wO + (size_t)l * 768 * 768,
            kROWS, 768, 768, o_b + (size_t)l * 768, xbuf, nullptr, 768, mvec + 1536, 4608, kN, 3);
        // h2 = LN(x)*(1+sc2)+sh2
        mod_ln<<<kROWS, 256, 0, stream>>>(xbuf, mvec, 3 * 768, 4 * 768, Hmod);
        // gelu(h2 @ m1_W + b) -> H3 (f16)
        gemm_f16<<<dim3(kROWS / 64, 3072 / 64), 128, 0, stream>>>(Hmod, wM1 + (size_t)l * 768 * 3072,
            kROWS, 3072, 768, m1_b + (size_t)l * 3072, nullptr, H3, 3072, nullptr, 0, 0, 2);
        // x += g2 * (H3 @ m2_W + b)
        gemm_f16<<<dim3(kROWS / 64, 768 / 64), 128, 0, stream>>>(H3, wM2 + (size_t)l * 3072 * 768,
            kROWS, 768, 3072, m2_b + (size_t)l * 768, xbuf, nullptr, 768, mvec + 3840, 4608, kN, 3);
    }

    // ---- output projection + pooling ----
    cvt_f16<<<cld((size_t)kROWS * 768, 256), 256, 0, stream>>>(xbuf, Hmod, (size_t)kROWS * 768);
    gemm_f16<<<dim3(kROWS / 64, 256 / 64), 128, 0, stream>>>(Hmod, wTout, kROWS, 256, 768,
        tout_b, S32, nullptr, 256, nullptr, 0, 0, 0);
    float* outp = (float*)d_out;
    float* AL = outp + (size_t)kB * kNR * kDA;   // atom_latent region
    ln_rows<<<kROWS, 256, 0, stream>>>(S32, 256, tout_ln_g, tout_ln_b, AL, nullptr, 256, 0);
    colsum_kernel<<<cld(kB * kNR, 256), 256, 0, stream>>>(a2t, csum);
    residue_reduce<<<kB * kNR, 256, 0, stream>>>(a2t, AL, csum, outp);
}
